// Net_78915729096910
// MI455X (gfx1250) — matrix operand order repaired and checked by reading, not compiled, not run on hardware
//
#include <hip/hip_runtime.h>

// ============================================================================
// PointNet++ segmentation net for MI455X (gfx1250, wave32, WMMA).
// All MLP GEMMs run on v_wmma_f32_16x16x32_f16 (f16 in, f32 accumulate).
// Weights converted once per launch to transposed, zero-K-padded f16.
// Input tiles are staged to LDS with GLOBAL_LOAD_ASYNC_TO_LDS_B128
// (ASYNCcnt / s_wait_asynccnt), the CDNA5 async copy path.
// SA max-aggregation uses atomicMax on float bits (valid: relu outputs >= 0,
// every query ball contains the query itself so cnt >= 1, buffer zeroed).
// Workspace use ~111 MB (dominant: 67 MB shared f16 gather buffer, reused
// across all six MLP stages since the pipeline is sequential on `stream`).
// ============================================================================

#define MLP_ASYNC_STAGE 1   // 0 -> plain vectorized uint4 staging copy

typedef _Float16 v16h __attribute__((ext_vector_type(16)));
typedef _Float16 v8h  __attribute__((ext_vector_type(8)));
typedef float    v8f  __attribute__((ext_vector_type(8)));

static constexpr int B_  = 16;
static constexpr int N_  = 4096;
static constexpr int S1_ = 820;
static constexpr int S2_ = 205;
static constexpr int K_  = 64;

// ---------------------------------------------------------------------------
// Weight convert: W[din][dout] f32 -> Wt[coutPad][kpad] f16 (transposed,
// zero padded) + padded f32 bias.  Transposed layout makes the WMMA
// B-fragment two contiguous 16-byte loads per lane.
// ---------------------------------------------------------------------------
__global__ void convw_kernel(const float* __restrict__ W, const float* __restrict__ bias,
                             int din, int dout, int kpad, int coutPad,
                             _Float16* __restrict__ wt, float* __restrict__ bpad)
{
    int i = blockIdx.x * blockDim.x + threadIdx.x;
    int total = kpad * coutPad;
    if (i < total) {
        int n = i / kpad, k = i - n * kpad;
        float v = (n < dout && k < din) ? W[(size_t)k * dout + n] : 0.f;
        wt[i] = (_Float16)v;
    }
    if (i < coutPad) bpad[i] = (i < dout) ? bias[i] : 0.f;
}

// ---------------------------------------------------------------------------
// Farthest point sampling: one 256-thread block per cloud, dmin in registers
// (16 pts/thread), argmax via LDS tree reduction. Start index 0, first-max
// tie-break (smallest index) to match jnp.argmax.
// ---------------------------------------------------------------------------
__global__ __launch_bounds__(256)
void fps_kernel(const float* __restrict__ pos, int n, int nsamp,
                int* __restrict__ idxOut, float* __restrict__ posOut)
{
    const int b = blockIdx.x;
    const float* p = pos + (size_t)b * n * 3;
    int*   io = idxOut + (size_t)b * nsamp;
    float* po = posOut + (size_t)b * nsamp * 3;
    const int tid = threadIdx.x;
    __shared__ float cur[3];
    __shared__ float sval[256];
    __shared__ int   sidx[256];
    float dmin[16];
#pragma unroll
    for (int k = 0; k < 16; ++k) dmin[k] = 1e30f;
    if (tid == 0) {
        io[0] = 0;
        cur[0] = p[0]; cur[1] = p[1]; cur[2] = p[2];
        po[0] = p[0]; po[1] = p[1]; po[2] = p[2];
    }
    __syncthreads();
    float cx = cur[0], cy = cur[1], cz = cur[2];
    for (int s = 1; s < nsamp; ++s) {
        float best = -1.f; int bj = n;
#pragma unroll
        for (int k = 0; k < 16; ++k) {
            int j = tid + (k << 8);
            if (j < n) {
                float dx = p[j*3] - cx, dy = p[j*3+1] - cy, dz = p[j*3+2] - cz;
                float d = dx*dx + dy*dy + dz*dz;
                if (d < dmin[k]) dmin[k] = d;
                float dm = dmin[k];
                if (dm > best) { best = dm; bj = j; }
            }
        }
        sval[tid] = best; sidx[tid] = bj;
        __syncthreads();
        for (int offr = 128; offr > 0; offr >>= 1) {
            if (tid < offr) {
                float ov = sval[tid + offr]; int oi = sidx[tid + offr];
                if (ov > sval[tid] || (ov == sval[tid] && oi < sidx[tid])) {
                    sval[tid] = ov; sidx[tid] = oi;
                }
            }
            __syncthreads();
        }
        if (tid == 0) {
            int j = sidx[0];
            io[s] = j;
            cur[0] = p[j*3]; cur[1] = p[j*3+1]; cur[2] = p[j*3+2];
            po[s*3] = cur[0]; po[s*3+1] = cur[1]; po[s*3+2] = cur[2];
        }
        __syncthreads();
        cx = cur[0]; cy = cur[1]; cz = cur[2];
    }
}

// ---------------------------------------------------------------------------
// Radius neighbors: first <= K points (ascending index) within r of each
// query (PyG semantics). One thread per query, sequential scan.
// ---------------------------------------------------------------------------
__global__ void radius_kernel(const float* __restrict__ posAll, const float* __restrict__ posQ,
                              int n, int S, float r2,
                              int* __restrict__ nbr, int* __restrict__ cnt)
{
    int q = blockIdx.x * blockDim.x + threadIdx.x;
    if (q >= B_ * S) return;
    int b = q / S;
    const float* pa = posAll + (size_t)b * n * 3;
    float qx = posQ[(size_t)q*3], qy = posQ[(size_t)q*3+1], qz = posQ[(size_t)q*3+2];
    int* out = nbr + (size_t)q * K_;
    int c = 0;
    for (int j = 0; j < n && c < K_; ++j) {
        float dx = pa[j*3] - qx, dy = pa[j*3+1] - qy, dz = pa[j*3+2] - qz;
        if (dx*dx + dy*dy + dz*dz <= r2) out[c++] = j;
    }
    cnt[q] = c;
    for (int k = c; k < K_; ++k) out[k] = 0;
}

// ---------------------------------------------------------------------------
// 3-NN (inverse squared distance) for knn_interpolate.
// ---------------------------------------------------------------------------
__global__ void knn3_kernel(const float* __restrict__ posT, const float* __restrict__ posS,
                            int T, int S, int* __restrict__ oIdx, float* __restrict__ oW)
{
    int t = blockIdx.x * blockDim.x + threadIdx.x;
    if (t >= B_ * T) return;
    int b = t / T;
    const float* ps = posS + (size_t)b * S * 3;
    float px = posT[(size_t)t*3], py = posT[(size_t)t*3+1], pz = posT[(size_t)t*3+2];
    float d0 = 3e30f, d1 = 3e30f, d2 = 3e30f; int i0 = 0, i1 = 0, i2 = 0;
    for (int j = 0; j < S; ++j) {
        float dx = ps[j*3] - px, dy = ps[j*3+1] - py, dz = ps[j*3+2] - pz;
        float d = dx*dx + dy*dy + dz*dz;
        if (d < d0)      { d2 = d1; i2 = i1; d1 = d0; i1 = i0; d0 = d; i0 = j; }
        else if (d < d1) { d2 = d1; i2 = i1; d1 = d; i1 = j; }
        else if (d < d2) { d2 = d; i2 = j; }
    }
    oIdx[t*3] = i0; oIdx[t*3+1] = i1; oIdx[t*3+2] = i2;
    oW[t*3]   = 1.f / fmaxf(d0, 1e-16f);
    oW[t*3+1] = 1.f / fmaxf(d1, 1e-16f);
    oW[t*3+2] = 1.f / fmaxf(d2, 1e-16f);
}

// ---------------------------------------------------------------------------
// Gather kernels: build zero-K-padded f16 input rows for the MLP kernel.
// ---------------------------------------------------------------------------
__global__ __launch_bounds__(32)
void gather_sa1_kernel(const float* __restrict__ x, const float* __restrict__ pos,
                       const float* __restrict__ p1, const int* __restrict__ nbr1,
                       const int* __restrict__ cnt1, _Float16* __restrict__ gin)
{
    int row = blockIdx.x;                   // [0, B*S1*K)
    int r = row & (K_ - 1); int qs = row >> 6;
    int b = qs / S1_;
    int tid = threadIdx.x;
    _Float16* o = gin + (size_t)row * 32;
    if (r >= cnt1[qs]) { o[tid] = (_Float16)0.f; return; }
    int j = nbr1[(size_t)qs * K_ + r];
    float v = 0.f;
    if (tid < 4)      v = x[((size_t)b*N_ + j)*4 + tid];
    else if (tid < 7) v = pos[((size_t)b*N_ + j)*3 + (tid-4)] - p1[(size_t)qs*3 + (tid-4)];
    o[tid] = (_Float16)v;
}

__global__ __launch_bounds__(64)
void gather_sa2_kernel(const float* __restrict__ x1, const float* __restrict__ p1,
                       const float* __restrict__ p2, const int* __restrict__ nbr2,
                       const int* __restrict__ cnt2, _Float16* __restrict__ gin)
{
    int row = blockIdx.x;                   // [0, B*S2*K)
    int r = row & (K_ - 1); int qs = row >> 6;
    int b = qs / S2_;
    bool valid = r < cnt2[qs];
    int j = valid ? nbr2[(size_t)qs * K_ + r] : 0;
    const float* xs = x1 + ((size_t)b*S1_ + j) * 128;
    const float* pj = p1 + ((size_t)b*S1_ + j) * 3;
    const float* pq = p2 + (size_t)qs * 3;
    _Float16* o = gin + (size_t)row * 160;
    for (int c = threadIdx.x; c < 160; c += 64) {
        float v = 0.f;
        if (valid) {
            if (c < 128)      v = xs[c];
            else if (c < 131) v = pj[c-128] - pq[c-128];
        }
        o[c] = (_Float16)v;
    }
}

__global__ __launch_bounds__(64)
void gather_sa3_kernel(const float* __restrict__ x2, const float* __restrict__ p2,
                       _Float16* __restrict__ gin)
{
    int row = blockIdx.x;                   // [0, B*S2)
    _Float16* o = gin + (size_t)row * 288;
    for (int c = threadIdx.x; c < 288; c += 64) {
        float v = 0.f;
        if (c < 256)      v = x2[(size_t)row*256 + c];
        else if (c < 259) v = p2[(size_t)row*3 + (c-256)];
        o[c] = (_Float16)v;
    }
}

__global__ void maxpool_kernel(const float* __restrict__ h3, float* __restrict__ xg)
{
    int i = blockIdx.x * blockDim.x + threadIdx.x;
    if (i >= B_ * 1024) return;
    int b = i >> 10, c = i & 1023;
    float m = -1e30f;
    for (int s = 0; s < S2_; ++s) m = fmaxf(m, h3[((size_t)b*S2_ + s)*1024 + c]);
    xg[i] = m;
}

__global__ __launch_bounds__(64)
void gather_fp1_kernel(const float* __restrict__ xg, const float* __restrict__ x2,
                       _Float16* __restrict__ gin)
{
    int row = blockIdx.x;                   // [0, B*S2)
    int b = row / S2_;
    _Float16* o = gin + (size_t)row * 1280;
    for (int c = threadIdx.x; c < 1280; c += 64) {
        float v = (c < 1024) ? xg[(size_t)b*1024 + c] : x2[(size_t)row*256 + (c-1024)];
        o[c] = (_Float16)v;
    }
}

__global__ __launch_bounds__(64)
void gather_fp2_kernel(const float* __restrict__ f1, const float* __restrict__ x1,
                       const int* __restrict__ ki, const float* __restrict__ kw,
                       _Float16* __restrict__ gin)
{
    int row = blockIdx.x;                   // [0, B*S1)
    int b = row / S1_;
    int i0 = ki[row*3], i1 = ki[row*3+1], i2 = ki[row*3+2];
    float w0 = kw[row*3], w1 = kw[row*3+1], w2 = kw[row*3+2];
    float inv = 1.f / (w0 + w1 + w2);
    const float* s0 = f1 + ((size_t)b*S2_ + i0) * 256;
    const float* s1 = f1 + ((size_t)b*S2_ + i1) * 256;
    const float* s2 = f1 + ((size_t)b*S2_ + i2) * 256;
    _Float16* o = gin + (size_t)row * 384;
    for (int c = threadIdx.x; c < 384; c += 64) {
        float v = (c < 256) ? (w0*s0[c] + w1*s1[c] + w2*s2[c]) * inv
                            : x1[(size_t)row*128 + (c-256)];
        o[c] = (_Float16)v;
    }
}

__global__ __launch_bounds__(64)
void gather_tail_kernel(const float* __restrict__ f2, const float* __restrict__ x,
                        const int* __restrict__ ki, const float* __restrict__ kw,
                        _Float16* __restrict__ gin)
{
    int row = blockIdx.x;                   // [0, B*N)
    int b = row / N_;
    int i0 = ki[row*3], i1 = ki[row*3+1], i2 = ki[row*3+2];
    float w0 = kw[row*3], w1 = kw[row*3+1], w2 = kw[row*3+2];
    float inv = 1.f / (w0 + w1 + w2);
    const float* s0 = f2 + ((size_t)b*S1_ + i0) * 128;
    const float* s1 = f2 + ((size_t)b*S1_ + i1) * 128;
    const float* s2 = f2 + ((size_t)b*S1_ + i2) * 128;
    _Float16* o = gin + (size_t)row * 160;
    for (int c = threadIdx.x; c < 160; c += 64) {
        float v = 0.f;
        if (c < 128)      v = (w0*s0[c] + w1*s1[c] + w2*s2[c]) * inv;
        else if (c < 132) v = x[(size_t)row*4 + (c-128)];
        o[c] = (_Float16)v;
    }
}

// ---------------------------------------------------------------------------
// Generic fused MLP on WMMA.  16 rows / workgroup, 4 waves split 16x16
// column tiles, up to 6 chained Linear+ReLU layers in LDS ping-pong (f16).
// Fragment layouts per CDNA5 ISA 7.12.2:
//   A (16x32 f16): lane M=lane%16, chunks K=[hi8,hi8+8) and [16+hi8,16+hi8+8)
//   B (32x16 f16): lane N=lane%16, same K chunks as A -> two 16B pieces of a row of
//                  the pre-transposed weight matrix
//   C/D (16x16 f32): lane N=lane%16, vgpr r -> M = r + 8*(lane>=16)
// Input tile staged to LDS via GLOBAL_LOAD_ASYNC_TO_LDS_B128: the tile is
// contiguous (row stride == ldIn) and dynamic LDS bufA sits at LDS offset 0,
// so the per-lane LDS address is just the byte offset of the 16B chunk.
// aggregate mode: final-layer relu outputs max-reduced over groups of 64
// rows via atomicMax on float bits (cnt-masked), buffer pre-zeroed.
// ---------------------------------------------------------------------------
struct LayerDesc { const _Float16* wt; const float* bias; int kpad; int cout; int relu; };
struct MlpDesc   { LayerDesc L[6]; int nLayers; };

__global__ __launch_bounds__(128)
void mlp_rows_kernel(const _Float16* __restrict__ gin, int ldIn,
                     float* __restrict__ gout, int ldOut, int coutLogical,
                     MlpDesc md, const int* __restrict__ cnt, int aggregate, int maxw)
{
    extern __shared__ _Float16 smem[];
    _Float16* bufA = smem;                       // LDS byte offset 0
    _Float16* bufB = smem + (size_t)16 * maxw;
    const int tid  = threadIdx.x;
    const int wave = tid >> 5;
    const int lane = tid & 31;
    const int M    = lane & 15;
    const int hi8  = (lane >> 4) << 3;        // 0 or 8
    const int row0 = blockIdx.x << 4;

#if MLP_ASYNC_STAGE
    {
        // 16 rows x ldIn halfs, contiguous: async-copy 16B chunks to LDS.
        const unsigned tileBytes = 16u * (unsigned)ldIn * 2u;   // mult of 1024
        const unsigned long long src =
            (unsigned long long)(const void*)(gin + (size_t)row0 * ldIn);
        for (unsigned o = (unsigned)tid * 16u; o < tileBytes; o += 128u * 16u) {
            unsigned lds = o;                    // bufA LDS offset == o
            unsigned long long ga = src + o;
            asm volatile("global_load_async_to_lds_b128 %0, %1, off"
                         :: "v"(lds), "v"(ga) : "memory");
        }
        asm volatile("s_wait_asynccnt 0x0" ::: "memory");
    }
#else
    {
        const unsigned chunks = 16u * (unsigned)ldIn / 8u;      // 8 halfs / uint4
        const uint4* src = (const uint4*)(gin + (size_t)row0 * ldIn);
        uint4* dst = (uint4*)bufA;
        for (unsigned i = tid; i < chunks; i += 128u) dst[i] = src[i];
    }
#endif
    __syncthreads();

    _Float16* cur = bufA; _Float16* nxt = bufB;
    for (int l = 0; l < md.nLayers; ++l) {
        const _Float16* wt = md.L[l].wt;
        const float* bias  = md.L[l].bias;
        const int kp   = md.L[l].kpad;
        const int co   = md.L[l].cout;
        const int relu = md.L[l].relu;
        const bool last = (l == md.nLayers - 1);
        const int ntiles = co >> 4;
        for (int t = wave; t < ntiles; t += 4) {
            const int n = (t << 4) + M;
            v8f acc;
            const float bv = bias[n];
#pragma unroll
            for (int r = 0; r < 8; ++r) acc[r] = bv;
            const _Float16* wrow = wt + (size_t)n * kp + hi8;  // hi8 = 0|8
            const _Float16* arow = cur + M * kp + hi8;
            for (int k0 = 0; k0 < kp; k0 += 32) {
                v8h alo = *(const v8h*)(arow + k0);
                v8h ahi = *(const v8h*)(arow + k0 + 16);
                v16h a, b;
#pragma unroll
                for (int e = 0; e < 8; ++e) { a[e] = alo[e]; a[8 + e] = ahi[e]; }
                v8h blo = *(const v8h*)(wrow + k0);
                v8h bhi = *(const v8h*)(wrow + k0 + 16);
#pragma unroll
                for (int e = 0; e < 8; ++e) { b[e] = blo[e]; b[8 + e] = bhi[e]; }
                acc = __builtin_amdgcn_wmma_f32_16x16x32_f16(
                    false, a, false, b, (short)0, acc, false, false);
            }
            if (!last) {
#pragma unroll
                for (int r = 0; r < 8; ++r) {
                    float v = acc[r];
                    if (relu) v = v > 0.f ? v : 0.f;
                    nxt[(r + hi8) * co + n] = (_Float16)v;
                }
            } else if (aggregate) {
#pragma unroll
                for (int r = 0; r < 8; ++r) {
                    float v = acc[r];
                    if (relu) v = v > 0.f ? v : 0.f;
                    const int row = row0 + r + hi8;
                    const int q = row >> 6;
                    if ((row & 63) < cnt[q])
                        atomicMax((unsigned int*)&gout[(size_t)q * ldOut + n],
                                  __float_as_uint(v));
                }
            } else {
#pragma unroll
                for (int r = 0; r < 8; ++r) {
                    float v = acc[r];
                    if (relu) v = v > 0.f ? v : 0.f;
                    if (n < coutLogical)
                        gout[(size_t)(row0 + r + hi8) * ldOut + n] = v;
                }
            }
        }
        __syncthreads();
        _Float16* tmp = cur; cur = nxt; nxt = tmp;
    }
}

// ===========================================================================
extern "C" void kernel_launch(void* const* d_in, const int* in_sizes, int n_in,
                              void* d_out, int out_size, void* d_ws, size_t ws_size,
                              hipStream_t stream)
{
    (void)in_sizes; (void)n_in; (void)out_size; (void)ws_size;
    const float* x   = (const float*)d_in[0];
    const float* pos = (const float*)d_in[1];
    // d_in[2] = batch (implied by layout), d_in[3] = force_vector (unused)
    float* outp = (float*)d_out;

    // 19 Linear layers, params at d_in[4 + 2l] / d_in[5 + 2l]
    struct LInfo { int din, dout, kpad, coutPad; };
    const LInfo LI[19] = {
        {7,64,32,64},       {64,64,64,64},      {64,128,64,128},       // sa1
        {131,128,160,128},  {128,128,128,128},  {128,256,128,256},     // sa2
        {259,256,288,256},  {256,512,256,512},  {512,1024,512,1024},   // sa3
        {1280,256,1280,256},{256,256,256,256},                         // fp1
        {384,256,384,256},  {256,128,256,128},                         // fp2
        {132,128,160,128},  {128,128,128,128},  {128,128,128,128},     // fp3
        {128,128,128,128},  {128,128,128,128},                         // lin
        {128,3,128,16},                                                // out
    };

    char* base = (char*)d_ws;
    size_t off = 0;
    auto alloc = [&](size_t bytes) -> char* {
        char* p = base + off;
        off = (off + bytes + 255) & ~(size_t)255;
        return p;
    };

    _Float16* wt[19]; float* bp[19];
    for (int l = 0; l < 19; ++l) {
        wt[l] = (_Float16*)alloc((size_t)LI[l].kpad * LI[l].coutPad * sizeof(_Float16));
        bp[l] = (float*)alloc((size_t)LI[l].coutPad * sizeof(float));
    }
    int*   idx1 = (int*)  alloc((size_t)B_*S1_*4);
    float* p1   = (float*)alloc((size_t)B_*S1_*3*4);
    int*   idx2 = (int*)  alloc((size_t)B_*S2_*4);
    float* p2   = (float*)alloc((size_t)B_*S2_*3*4);
    int*   nbr1 = (int*)  alloc((size_t)B_*S1_*K_*4);
    int*   cnt1 = (int*)  alloc((size_t)B_*S1_*4);
    int*   nbr2 = (int*)  alloc((size_t)B_*S2_*K_*4);
    int*   cnt2 = (int*)  alloc((size_t)B_*S2_*4);
    float* x1   = (float*)alloc((size_t)B_*S1_*128*4);
    float* x2   = (float*)alloc((size_t)B_*S2_*256*4);
    float* h3   = (float*)alloc((size_t)B_*S2_*1024*4);
    float* xg   = (float*)alloc((size_t)B_*1024*4);
    float* f1   = (float*)alloc((size_t)B_*S2_*256*4);
    float* f2   = (float*)alloc((size_t)B_*S1_*128*4);
    int*   kAi  = (int*)  alloc((size_t)B_*S1_*3*4);
    float* kAw  = (float*)alloc((size_t)B_*S1_*3*4);
    int*   kBi  = (int*)  alloc((size_t)B_*N_*3*4);
    float* kBw  = (float*)alloc((size_t)B_*N_*3*4);
    // shared gather region, sized for the largest stage (SA2: 209920 x 160 f16)
    _Float16* gin = (_Float16*)alloc((size_t)B_*S2_*K_*160*sizeof(_Float16));

    // ---- weight conversion (transpose + f16 + K/N padding) ----
    for (int l = 0; l < 19; ++l) {
        const float* W  = (const float*)d_in[4 + 2*l];
        const float* bb = (const float*)d_in[5 + 2*l];
        int total = LI[l].kpad * LI[l].coutPad;
        int thr = total > LI[l].coutPad ? total : LI[l].coutPad;
        convw_kernel<<<(thr + 255) / 256, 256, 0, stream>>>(
            W, bb, LI[l].din, LI[l].dout, LI[l].kpad, LI[l].coutPad, wt[l], bp[l]);
    }

    auto LD = [&](int l) -> LayerDesc {
        return LayerDesc{ wt[l], bp[l], LI[l].kpad, LI[l].coutPad, 1 };
    };

    // ---- SA1: fps -> radius -> PointConv(max) ----
    fps_kernel<<<B_, 256, 0, stream>>>(pos, N_, S1_, idx1, p1);
    radius_kernel<<<(B_*S1_ + 255)/256, 256, 0, stream>>>(pos, p1, N_, S1_, 0.04f, nbr1, cnt1);
    gather_sa1_kernel<<<B_*S1_*K_, 32, 0, stream>>>(x, pos, p1, nbr1, cnt1, gin);
    hipMemsetAsync(x1, 0, (size_t)B_*S1_*128*4, stream);
    {
        MlpDesc md{}; md.nLayers = 3; md.L[0] = LD(0); md.L[1] = LD(1); md.L[2] = LD(2);
        mlp_rows_kernel<<<(B_*S1_*K_)/16, 128, (size_t)2*16*128*2, stream>>>(
            gin, 32, x1, 128, 128, md, cnt1, 1, 128);
    }

    // ---- SA2 ----
    fps_kernel<<<B_, 256, 0, stream>>>(p1, S1_, S2_, idx2, p2);
    radius_kernel<<<(B_*S2_ + 255)/256, 256, 0, stream>>>(p1, p2, S1_, S2_, 0.16f, nbr2, cnt2);
    gather_sa2_kernel<<<B_*S2_*K_, 64, 0, stream>>>(x1, p1, p2, nbr2, cnt2, gin);
    hipMemsetAsync(x2, 0, (size_t)B_*S2_*256*4, stream);
    {
        MlpDesc md{}; md.nLayers = 3; md.L[0] = LD(3); md.L[1] = LD(4); md.L[2] = LD(5);
        mlp_rows_kernel<<<(B_*S2_*K_)/16, 128, (size_t)2*16*256*2, stream>>>(
            gin, 160, x2, 256, 256, md, cnt2, 1, 256);
    }

    // ---- GlobalSA (sa3) + global max pool ----
    gather_sa3_kernel<<<B_*S2_, 64, 0, stream>>>(x2, p2, gin);
    {
        MlpDesc md{}; md.nLayers = 3; md.L[0] = LD(6); md.L[1] = LD(7); md.L[2] = LD(8);
        mlp_rows_kernel<<<(B_*S2_)/16, 128, (size_t)2*16*1024*2, stream>>>(
            gin, 288, h3, 1024, 1024, md, nullptr, 0, 1024);
    }
    maxpool_kernel<<<(B_*1024 + 255)/256, 256, 0, stream>>>(h3, xg);

    // ---- FP1: broadcast global feature + skip x2 ----
    gather_fp1_kernel<<<B_*S2_, 64, 0, stream>>>(xg, x2, gin);
    {
        MlpDesc md{}; md.nLayers = 2; md.L[0] = LD(9); md.L[1] = LD(10);
        mlp_rows_kernel<<<(B_*S2_)/16, 128, (size_t)2*16*1280*2, stream>>>(
            gin, 1280, f1, 256, 256, md, nullptr, 0, 1280);
    }

    // ---- FP2: knn-interp S2 -> S1, skip x1 ----
    knn3_kernel<<<(B_*S1_ + 255)/256, 256, 0, stream>>>(p1, p2, S1_, S2_, kAi, kAw);
    gather_fp2_kernel<<<B_*S1_, 64, 0, stream>>>(f1, x1, kAi, kAw, gin);
    {
        MlpDesc md{}; md.nLayers = 2; md.L[0] = LD(11); md.L[1] = LD(12);
        mlp_rows_kernel<<<(B_*S1_)/16, 128, (size_t)2*16*384*2, stream>>>(
            gin, 384, f2, 128, 128, md, nullptr, 0, 384);
    }

    // ---- FP3 + lin + out, fused 6-layer tail straight into d_out ----
    knn3_kernel<<<(B_*N_ + 255)/256, 256, 0, stream>>>(pos, p1, N_, S1_, kBi, kBw);
    gather_tail_kernel<<<B_*N_, 64, 0, stream>>>(f2, x, kBi, kBw, gin);
    {
        MlpDesc md{}; md.nLayers = 6;
        md.L[0] = LD(13); md.L[1] = LD(14); md.L[2] = LD(15);
        md.L[3] = LD(16); md.L[4] = LD(17);
        md.L[5] = LD(18); md.L[5].relu = 0;          // final linear: no relu
        mlp_rows_kernel<<<(B_*N_)/16, 128, (size_t)2*16*160*2, stream>>>(
            gin, 160, outp, 3, 3, md, nullptr, 0, 160);
    }
}
